// PosteriorRNN_64037962383370
// MI455X (gfx1250) — matrix operand order repaired and checked by reading, not compiled, not run on hardware
//
#include <hip/hip_runtime.h>

// ---------------- problem constants ----------------
#define XD     64          // x_dim
#define HID    256         // hidden
#define G4     1024        // 4*HID (gates i,f,g,o stacked)
#define KTOT   320         // XD + HID : fused GEMM K
#define BB     64          // batch
#define TT     2048        // timesteps
#define MROWS  16          // batch rows per workgroup (WMMA M)
#define NWG    (BB / MROWS)          // 4 workgroups, fully independent
#define KSTEPS (KTOT / 32)           // 10 WMMA K-slices per step (2 x-slices + 8 h-slices)

typedef __attribute__((ext_vector_type(16))) _Float16 v16h;
typedef __attribute__((ext_vector_type(8)))  _Float16 v8h;
typedef __attribute__((ext_vector_type(8)))  float    v8f;

// ---------------- prep: fp32 x -> f16 copy ----------------
__global__ void cvt_x_kernel(const float* __restrict__ x, _Float16* __restrict__ xh, int n) {
  int i = blockIdx.x * blockDim.x + threadIdx.x;
  if (i < n) xh[i] = (_Float16)x[i];
}

// ---------------- prep: swizzle W = [w_ih | w_hh] into per-lane WMMA B-fragments ----
// wsw layout: frag(tile, kk) is 32 lanes x 16 halves contiguous:
//   wsw[((tile*KSTEPS + kk)*32 + lane)*16 + s]  = W[tile*16 + lane%16][kk*32 + 8*(lane/16) + s%8 + 16*(s/8)]
__global__ void prep_w_kernel(const float* __restrict__ w_ih, const float* __restrict__ w_hh,
                              _Float16* __restrict__ wsw) {
  int id = blockIdx.x * blockDim.x + threadIdx.x;   // 64 tiles * 10 kk * 32 lanes * 16 n
  if (id >= (G4 / 16) * KSTEPS * 32 * 16) return;
  int nh   = id & 15;
  int lane = (id >> 4) & 31;
  int rest = id >> 9;                 // tile*KSTEPS + kk
  int kk   = rest % KSTEPS;
  int tile = rest / KSTEPS;
  int k = kk * 32 + ((lane >> 4) << 3) + (nh & 7) + ((nh >> 3) << 4);  // fused K index (0..319)
  int n = tile * 16 + (lane & 15);    // gate-output row (0..1023)
  float v = (k < XD) ? w_ih[n * XD + k] : w_hh[n * HID + (k - XD)];
  wsw[id] = (_Float16)v;
}

// ---------------- gate math (trans ops co-execute with WMMA) ----------------
__device__ __forceinline__ float fsig(float x) {
  return __builtin_amdgcn_rcpf(1.0f + __expf(-x));
}
__device__ __forceinline__ float ftanh(float x) {
  return 1.0f - 2.0f * __builtin_amdgcn_rcpf(__expf(2.0f * x) + 1.0f);
}

// ---------------- main: sequential scan, one WGP per 16 batch rows ----------------
__global__ void __launch_bounds__(256, 1)
lstm_scan_kernel(const _Float16* __restrict__ xh,   // (B, T, XD) f16
                 const _Float16* __restrict__ wsw,  // swizzled fused weights, f16
                 const float*    __restrict__ bias, // (4H,) f32
                 float*          __restrict__ out)  // (B, T, H) f32
{
  __shared__ _Float16 ldsH[MROWS * HID];      // h(t-1), row stride 256 halves
  __shared__ _Float16 ldsX[2][MROWS * XD];    // double-buffered x_t, row stride 64 halves

  const int tid  = threadIdx.x;
  const int lane = tid & 31;
  const int wave = tid >> 5;                // 0..7 ; owns hidden blocks {wave, wave+8}
  const int m0   = blockIdx.x * MROWS;      // batch row base
  const int Nc   = lane & 15;               // N column within 16x16 C/D tile
  const int mhi  = (lane >> 4) * 8;         // M base within C/D tile

  // per-tile bias, loaded once (tile j = (j&3)*16 + wave + (j>>2)*8)
  float bv[8];
#pragma unroll
  for (int j = 0; j < 8; ++j)
    bv[j] = bias[((j & 3) * 16 + (wave + (j >> 2) * 8)) * 16 + Nc];

  // c state lives in VGPRs in C/D tile layout: cst[bi][e] = c[mhi+e][hb*16+Nc]
  v8f cst[2];
#pragma unroll
  for (int e = 0; e < 8; ++e) { cst[0][e] = 0.0f; cst[1][e] = 0.0f; }

  // init LDS: h(=0)
  for (int i = tid; i < MROWS * HID; i += 256) ldsH[i] = (_Float16)0.0f;

  // async global->LDS copy of x(0) into slot 0 (8 bytes/lane, ASYNCcnt-tracked)
  const int xm  = tid >> 4;                 // x-staging: row and 4-half column
  const int xc4 = (tid & 15) * 4;
  {
    unsigned ldso = (unsigned)(uintptr_t)&ldsX[0][xm * XD + xc4];
    const _Float16* g = xh + (size_t)(m0 + xm) * TT * XD + xc4;
    asm volatile("global_load_async_to_lds_b64 %0, %1, off"
                 :: "v"(ldso), "v"(g) : "memory");
  }

#pragma unroll 1
  for (int t = 0; t < TT; ++t) {
    // drain this wave's async x copy, then publish LDS to all waves
    asm volatile("s_wait_asynccnt 0x0" ::: "memory");
    __syncthreads();   // x(t) slot + h(t-1) complete

    // issue async copy of x(t+1) into the inactive slot; a whole GEMM+gate
    // phase hides its latency, drained at the next loop-top wait
    if (t + 1 < TT) {
      unsigned ldso = (unsigned)(uintptr_t)&ldsX[(t + 1) & 1][xm * XD + xc4];
      const _Float16* g = xh + (size_t)(m0 + xm) * TT * XD + (size_t)(t + 1) * XD + xc4;
      asm volatile("global_load_async_to_lds_b64 %0, %1, off"
                   :: "v"(ldso), "v"(g) : "memory");
    }

    // Opaque zero offset: prevents LICM from hoisting the (loop-invariant)
    // weight-fragment loads out of the 2048-step loop and spilling 640 VGPRs
    // of weights to scratch. Weights stay L2-resident, re-streamed per step.
    size_t wofs;
    asm volatile("s_mov_b64 %0, 0" : "=s"(wofs));
    const _Float16* fb = wsw + wofs + (size_t)lane * 16;  // per-lane fragment base
    auto fragp = [&](int j, int kk) -> const v16h* {
      int tile = (j & 3) * 16 + (wave + (j >> 2) * 8);
      return (const v16h*)(fb + (size_t)(tile * KSTEPS + kk) * 512);
    };

    // accumulators start at bias
    v8f acc[8];
#pragma unroll
    for (int j = 0; j < 8; ++j) {
      v8f a;
#pragma unroll
      for (int e = 0; e < 8; ++e) a[e] = bv[j];
      acc[j] = a;
    }

    // fused GEMM: z(16x1024) = [x_t | h] (16x320) * W^T
    // B fragments double-buffered in registers: loads for slice kk+1 are in
    // flight while the 8 WMMAs of slice kk execute (hides L2 latency).
    const _Float16* xslot = &ldsX[t & 1][0];
    v16h bfr[2][8];
#pragma unroll
    for (int j = 0; j < 8; ++j) bfr[0][j] = *fragp(j, 0);

#pragma unroll
    for (int kk = 0; kk < KSTEPS; ++kk) {
      // A fragment (16-bit A 16x32 layout): lanes 0-15 -> M=lane, K 0..7 & 16..23;
      // lanes 16-31 -> M=lane-16, K 8..15 & 24..31
      const _Float16* abuf;
      int abase;
      if (kk < 2) {  // x part
        abuf  = xslot;
        abase = (lane & 15) * XD + kk * 32 + ((lane >> 4) << 3);
      } else {       // h part
        abuf  = &ldsH[0];
        abase = (lane & 15) * HID + (kk - 2) * 32 + ((lane >> 4) << 3);
      }
      v8h lo = *(const v8h*)&abuf[abase];
      v8h hi = *(const v8h*)&abuf[abase + 16];
      v16h afrag;
#pragma unroll
      for (int i = 0; i < 8; ++i) { afrag[i] = lo[i]; afrag[i + 8] = hi[i]; }

      if (kk + 1 < KSTEPS) {   // prefetch next K-slice into the other buffer
#pragma unroll
        for (int j = 0; j < 8; ++j) bfr[(kk + 1) & 1][j] = *fragp(j, kk + 1);
      }

#pragma unroll
      for (int j = 0; j < 8; ++j) {
        acc[j] = __builtin_amdgcn_wmma_f32_16x16x32_f16(
            false, afrag, false, bfr[kk & 1][j], (short)0, acc[j], false, false);
      }
    }

    __syncthreads();   // all LDS reads of this step done before h is overwritten

    // gates -> (c,h) update; write h back to LDS and to global output
#pragma unroll
    for (int bi = 0; bi < 2; ++bi) {
      int hb = wave + bi * 8;
      v8f zi = acc[bi * 4 + 0], zf = acc[bi * 4 + 1];
      v8f zg = acc[bi * 4 + 2], zo = acc[bi * 4 + 3];
#pragma unroll
      for (int e = 0; e < 8; ++e) {
        float iv = fsig(zi[e]);
        float fv = fsig(zf[e]);
        float gv = ftanh(zg[e]);
        float ov = fsig(zo[e]);
        float cv = fv * cst[bi][e] + iv * gv;
        cst[bi][e] = cv;
        float hv = ov * ftanh(cv);
        int M = mhi + e;
        ldsH[M * HID + hb * 16 + Nc] = (_Float16)hv;
        out[(size_t)(m0 + M) * TT * HID + (size_t)t * HID + hb * 16 + Nc] = hv;
      }
    }
  }
}

// ---------------- launch ----------------
extern "C" void kernel_launch(void* const* d_in, const int* in_sizes, int n_in,
                              void* d_out, int out_size, void* d_ws, size_t ws_size,
                              hipStream_t stream) {
  const float* x_seq = (const float*)d_in[0];   // (B, T, XD)
  const float* w_ih  = (const float*)d_in[1];   // (4H, XD)
  const float* w_hh  = (const float*)d_in[2];   // (4H, H)
  const float* bias  = (const float*)d_in[3];   // (4H,)
  float* out = (float*)d_out;                   // (B, T, H)

  // workspace: xh (B*T*XD f16 = 16 MB) | wsw (4H*KTOT f16 = 640 KB)
  _Float16* xh  = (_Float16*)d_ws;
  _Float16* wsw = (_Float16*)((char*)d_ws + (size_t)BB * TT * XD * sizeof(_Float16));

  int nx = BB * TT * XD;
  cvt_x_kernel<<<(nx + 255) / 256, 256, 0, stream>>>(x_seq, xh, nx);

  int nw = (G4 / 16) * KSTEPS * 32 * 16;        // 327680
  prep_w_kernel<<<(nw + 255) / 256, 256, 0, stream>>>(w_ih, w_hh, wsw);

  lstm_scan_kernel<<<NWG, 256, 0, stream>>>(xh, wsw, bias, out);
}
